// STLN_58531814310353
// MI455X (gfx1250) — compile-verified, matrix operand order checked by reading, not run on hardware
//
#include <hip/hip_runtime.h>
#include <hip/hip_bf16.h>

// ---------------------------------------------------------------------------
// STLN for MI455X (gfx1250): bf16 WMMA GEMM path with LDS operand staging.
// B=32, T=63, NBONES=64, H=64, NGATES=9, 3 recurrent steps.
// ---------------------------------------------------------------------------

#define DEVI __device__ __forceinline__

typedef __attribute__((ext_vector_type(16))) __bf16 v16bf;
typedef __attribute__((ext_vector_type(8)))  float  v8f;

union FragA {
    v16bf v;
    uint4 q[2];
};

constexpr int B_  = 32;
constexpr int T_  = 63;
constexpr int NB_ = 64;
constexpr int H_  = 64;
constexpr size_t S_H = (size_t)B_ * T_ * NB_ * H_;   // 8,257,536 elements

DEVI v8f wmma_bf16(v16bf a, v16bf b, v8f c) {
    return __builtin_amdgcn_wmma_f32_16x16x32_bf16(
        /*neg_a=*/false, a, /*neg_b=*/false, b,
        /*c_mod=*/(short)0, c, /*reuse_a=*/false, /*reuse_b=*/false);
}

// A fragment (16x32 bf16, MxK): lane L holds row m=L&15; group g=L>>4 holds
// K = {32*kstep + 8g + 0..7} and {32*kstep + 16 + 8g + 0..7}: two 16B runs.
DEVI FragA load_row_frag(const __bf16* rowptr, int kstep, int group) {
    FragA f;
    const __bf16* p0 = rowptr + kstep * 32 + group * 8;
    f.q[0] = *reinterpret_cast<const uint4*>(p0);
    f.q[1] = *reinterpret_cast<const uint4*>(p0 + 16);
    return f;
}

// Same fragment gather but from an LDS-resident 64-element row.
DEVI FragA lds_row_frag(const __bf16* rowptr, int kstep, int group) {
    FragA f;
    const __bf16* p0 = rowptr + kstep * 32 + group * 8;
    f.q[0] = *reinterpret_cast<const uint4*>(p0);
    f.q[1] = *reinterpret_cast<const uint4*>(p0 + 16);
    return f;
}

// B fragment from pre-packed buffer: 512 bf16 per fragment, 16 per lane (32B).
DEVI v16bf load_frag_pk(const __bf16* pk, size_t fragIdx, int lane) {
    FragA f;
    const __bf16* p = pk + fragIdx * 512 + (size_t)lane * 16;
    f.q[0] = *reinterpret_cast<const uint4*>(p);
    f.q[1] = *reinterpret_cast<const uint4*>(p + 8);
    return f.v;
}

DEVI float sigm(float x) { return 1.0f / (1.0f + __expf(-x)); }

// ---------------------------------------------------------------------------
// Pack weights_in (4096x4096 f32) into WMMA B-fragment layout, bf16.
// fragIdx = nt16*128 + kstep.  One wave per fragment.
// ---------------------------------------------------------------------------
__global__ void stln_pack_win(const float* __restrict__ W,
                              __hip_bfloat16* __restrict__ pk_) {
    __bf16* pk = reinterpret_cast<__bf16*>(pk_);
    int gid  = blockIdx.x * blockDim.x + threadIdx.x;
    int wid  = gid >> 5;
    int lane = gid & 31;
    if (wid >= 256 * 128) return;
    int nt16  = wid >> 7;    // 0..255
    int kstep = wid & 127;   // 0..127
    int n     = nt16 * 16 + (lane & 15);
    int group = lane >> 4;
    FragA f;
#pragma unroll
    for (int j = 0; j < 16; ++j) {
        int klocal = (j < 8) ? (8 * group + j) : (16 + 8 * group + (j - 8));
        int k = kstep * 32 + klocal;
        f.v[j] = (__bf16)W[(size_t)k * 4096 + n];
    }
    __bf16* dst = pk + (size_t)wid * 512 + (size_t)lane * 16;
    *reinterpret_cast<uint4*>(dst)     = f.q[0];
    *reinterpret_cast<uint4*>(dst + 8) = f.q[1];
}

// ---------------------------------------------------------------------------
// Pack the 63 small 64x64 matrices (U[9], Wt slices [27], Ws[9], Zt[9],
// Zs[ZS_IDX][9]) into B-fragment layout.  One wave per (mat, frag).
// ---------------------------------------------------------------------------
__global__ void stln_pack_small(const float* __restrict__ U,
                                const float* __restrict__ Wt,
                                const float* __restrict__ Ws,
                                const float* __restrict__ Zt,
                                const float* __restrict__ Zs,
                                __hip_bfloat16* __restrict__ pk_) {
    __bf16* pk = reinterpret_cast<__bf16*>(pk_);
    int gid  = blockIdx.x * blockDim.x + threadIdx.x;
    int wid  = gid >> 5;
    int lane = gid & 31;
    if (wid >= 63 * 8) return;
    int mat  = wid >> 3;
    int frag = wid & 7;          // nt*2 + kstep
    int nt   = frag >> 1;
    int kstep = frag & 1;

    const float* src;
    if (mat < 9) {
        src = U + (size_t)mat * 4096;
    } else if (mat < 36) {
        int tt = mat - 9, g = tt / 3, s = tt % 3;
        src = Wt + (size_t)g * 12288 + (size_t)s * 4096;   // Wt[g][s*64 .. ]
    } else if (mat < 45) {
        src = Ws + (size_t)(mat - 36) * 4096;
    } else if (mat < 54) {
        src = Zt + (size_t)(mat - 45) * 4096;
    } else {
        const int zs_idx[9] = {0, 1, 1, 3, 4, 5, 6, 7, 8};
        src = Zs + (size_t)zs_idx[mat - 54] * 4096;
    }

    int n     = nt * 16 + (lane & 15);
    int group = lane >> 4;
    FragA f;
#pragma unroll
    for (int j = 0; j < 16; ++j) {
        int klocal = (j < 8) ? (8 * group + j) : (16 + 8 * group + (j - 8));
        int k = kstep * 32 + klocal;
        f.v[j] = (__bf16)src[k * 64 + n];
    }
    __bf16* dst = pk + (size_t)mat * 4096 + (size_t)frag * 512 + (size_t)lane * 16;
    *reinterpret_cast<uint4*>(dst)     = f.q[0];
    *reinterpret_cast<uint4*>(dst + 8) = f.q[1];
}

// ---------------------------------------------------------------------------
// f32 -> bf16 elementwise convert (encoder inputs)
// ---------------------------------------------------------------------------
__global__ void stln_cvt_bf16(const float* __restrict__ src,
                              __hip_bfloat16* __restrict__ dst_, int n) {
    __bf16* dst = reinterpret_cast<__bf16*>(dst_);
    int i = blockIdx.x * blockDim.x + threadIdx.x;
    if (i < n) dst[i] = (__bf16)src[i];
}

// ---------------------------------------------------------------------------
// Input GEMM: h = enc(2016x4096) @ Win(4096x4096) + bias.
// Block = 128 threads (4 waves), all sharing one 64-col strip (nt64).
// M = 128 rows/block (32/wave).  B fragments staged per K-step into a
// double-buffered 8KB LDS tile -> 8x less L2 traffic on weights.
// ---------------------------------------------------------------------------
__global__ void __launch_bounds__(128)
stln_gemm_in(const __hip_bfloat16* __restrict__ A_,
             const __hip_bfloat16* __restrict__ Bpk_,
             const float* __restrict__ bias,
             float* __restrict__ c0,
             __hip_bfloat16* __restrict__ pbf_) {
    const __bf16* A   = reinterpret_cast<const __bf16*>(A_);
    const __bf16* Bpk = reinterpret_cast<const __bf16*>(Bpk_);
    __bf16* pbf       = reinterpret_cast<__bf16*>(pbf_);

    __shared__ __align__(16) __bf16 bsh[2][4][512];   // [buf][frag][512] = 8KB

    int mb    = blockIdx.x >> 6;        // 0..15  (M block of 128 rows)
    int nt64  = blockIdx.x & 63;        // 0..63  (64-col strip)
    int tid   = threadIdx.x;
    int w     = tid >> 5;               // wave 0..3
    int lane  = tid & 31;
    int m     = lane & 15, group = lane >> 4;

    int rowbase = mb * 128 + w * 32;
    const __bf16* arow0 = A + (size_t)(rowbase + m) * 4096;
    const __bf16* arow1 = A + (size_t)(rowbase + 16 + m) * 4096;

    // Stage helper: thread copies 32B of fragment j = tid>>5 for K-step k.
    int sj = tid >> 5;                  // frag index 0..3 (one wave per frag)
    const size_t sfb = (size_t)(nt64 * 4 + sj) * 128;
    auto stage = [&](int buf, int k) {
        const __bf16* src = Bpk + (sfb + k) * 512 + (size_t)lane * 16;
        uint4 q0 = *reinterpret_cast<const uint4*>(src);
        uint4 q1 = *reinterpret_cast<const uint4*>(src + 8);
        *reinterpret_cast<uint4*>(&bsh[buf][sj][lane * 16])     = q0;
        *reinterpret_cast<uint4*>(&bsh[buf][sj][lane * 16 + 8]) = q1;
    };

    v8f acc0[4] = {v8f{}, v8f{}, v8f{}, v8f{}};
    v8f acc1[4] = {v8f{}, v8f{}, v8f{}, v8f{}};

    stage(0, 0);
    __syncthreads();
    for (int k = 0; k < 128; ++k) {
        int buf = k & 1;
        stage(buf ^ 1, k + 1);          // k=127 stages 1KB past strip: still in ws
        __builtin_prefetch(arow0 + k * 32 + 256, 0, 1);
        FragA a0 = load_row_frag(arow0, k, group);
        FragA a1 = load_row_frag(arow1, k, group);
#pragma unroll
        for (int j = 0; j < 4; ++j) {
            FragA bf;
            bf.q[0] = *reinterpret_cast<const uint4*>(&bsh[buf][j][lane * 16]);
            bf.q[1] = *reinterpret_cast<const uint4*>(&bsh[buf][j][lane * 16 + 8]);
            acc0[j] = wmma_bf16(a0.v, bf.v, acc0[j]);
            acc1[j] = wmma_bf16(a1.v, bf.v, acc1[j]);
        }
        __syncthreads();
    }

#pragma unroll
    for (int j = 0; j < 4; ++j) {
        int col = nt64 * 64 + j * 16 + (lane & 15);
        float bv = bias[col];
#pragma unroll
        for (int r = 0; r < 8; ++r) {
            int row0 = rowbase + r + 8 * group;
            int row1 = row0 + 16;
            if (row0 < B_ * T_) {
                float v = acc0[j][r] + bv;
                size_t o = (size_t)row0 * 4096 + col;
                c0[o] = v; pbf[o] = (__bf16)v;
            }
            if (row1 < B_ * T_) {
                float v = acc1[j][r] + bv;
                size_t o = (size_t)row1 * 4096 + col;
                c0[o] = v; pbf[o] = (__bf16)v;
            }
        }
    }
}

// ---------------------------------------------------------------------------
// Means: g_t = mean over T (per b,n,h);  g_s = mean over bones (per b,t,h).
// ---------------------------------------------------------------------------
__global__ void stln_mean_t(const float* __restrict__ h, float* __restrict__ cgt,
                            __hip_bfloat16* __restrict__ gtb_) {
    __bf16* gtb = reinterpret_cast<__bf16*>(gtb_);
    int idx = blockIdx.x * blockDim.x + threadIdx.x;
    if (idx >= B_ * NB_ * H_) return;
    int k = idx & 63, n = (idx >> 6) & 63, b = idx >> 12;
    const float* p = h + (size_t)b * T_ * 4096 + n * 64 + k;
    float s = 0.f;
    for (int t = 0; t < T_; ++t) s += p[(size_t)t * 4096];
    s *= (1.0f / (float)T_);
    cgt[idx] = s;
    gtb[idx] = (__bf16)s;
}

__global__ void stln_mean_s(const float* __restrict__ h, float* __restrict__ cgs,
                            __hip_bfloat16* __restrict__ gsb_) {
    __bf16* gsb = reinterpret_cast<__bf16*>(gsb_);
    int idx = blockIdx.x * blockDim.x + threadIdx.x;
    if (idx >= B_ * T_ * H_) return;
    int k = idx & 63, bt = idx >> 6;     // bt = b*T + t
    const float* p = h + (size_t)bt * 4096 + k;
    float s = 0.f;
    for (int n = 0; n < NB_; ++n) s += p[n * 64];
    s *= (1.0f / (float)NB_);
    cgs[idx] = s;
    gsb[idx] = (__bf16)s;
}

// ---------------------------------------------------------------------------
// One recurrent step.  Block = 288 threads = 9 waves = 9 gates, covering one
// full (b,t) slice (64 bones x 64 hidden).  The 6 shifted A operand planes
// (p, h_tb, h, h_ta, h_sb, g_t) are staged once into LDS (48KB) and re-read
// per gate via ds_load; B fragments are reused across 4 M-tiles (4x cut).
// pre is produced/consumed per 16-column chunk via a 36KB LDS buffer.
// Per wave: 4nt x 2k x 4mt x 7 = 224 WMMAs.
// ---------------------------------------------------------------------------
__global__ void __launch_bounds__(288)
stln_step(const __hip_bfloat16* __restrict__ pbf_,
          const __hip_bfloat16* __restrict__ hin_,
          const float* __restrict__ c_in,
          const __hip_bfloat16* __restrict__ spk_,
          const float* __restrict__ bmat,
          const __hip_bfloat16* __restrict__ gtb_,
          const __hip_bfloat16* __restrict__ gsb_,
          const float* __restrict__ cgt,
          const float* __restrict__ cgs,
          float* __restrict__ c_out,
          __hip_bfloat16* __restrict__ hout_bf_,
          float* __restrict__ hout_f32) {
    const __bf16* pbf = reinterpret_cast<const __bf16*>(pbf_);
    const __bf16* hin = reinterpret_cast<const __bf16*>(hin_);
    const __bf16* spk = reinterpret_cast<const __bf16*>(spk_);
    const __bf16* gtb = reinterpret_cast<const __bf16*>(gtb_);
    const __bf16* gsb = reinterpret_cast<const __bf16*>(gsb_);
    __bf16* hout_bf = reinterpret_cast<__bf16*>(hout_bf_);

    // A operand planes: 0=p 1=h_tb 2=h 3=h_ta 4=h_sb 5=g_t   (6 x 8KB = 48KB)
    __shared__ __align__(16) __bf16 apl[6][64][64];
    // pre chunk: all 9 gates for one 16-col n-tile                   (36KB)
    __shared__ float pre_s[9][64][16];

    int bt = blockIdx.x;                 // 0..2015  = b*T + t
    int t  = bt % T_;
    int b  = bt / T_;

    int tid   = threadIdx.x;
    int g     = tid >> 5;                // gate 0..8
    int lane  = tid & 31;
    int m     = lane & 15;
    int group = lane >> 4;

    // ---- Stage A planes (all 288 threads; 384 rows x 8 quads) ----
    {
        const size_t rb = (size_t)bt * 64;
        for (int q = tid; q < 6 * 64 * 8; q += 288) {
            int plane = q >> 9;          // /512
            int rem   = q & 511;
            int r     = rem >> 3;
            int quad  = rem & 7;
            const __bf16* src = nullptr;
            switch (plane) {
                case 0: src = pbf + (rb + r) * 64; break;
                case 1: src = (t > 0)      ? hin + (rb - 64 + r) * 64 : nullptr; break;
                case 2: src = hin + (rb + r) * 64; break;
                case 3: src = (t < T_ - 1) ? hin + (rb + 64 + r) * 64 : nullptr; break;
                case 4: src = (r > 0)      ? hin + (rb + r - 1) * 64  : nullptr; break;
                case 5: src = gtb + ((size_t)b * 64 + r) * 64; break;
            }
            uint4 qv = make_uint4(0u, 0u, 0u, 0u);
            if (src) qv = *reinterpret_cast<const uint4*>(src + quad * 8);
            *reinterpret_cast<uint4*>(&apl[plane][r][quad * 8]) = qv;
        }
    }
    __syncthreads();

    // g_s row is identical for every bone: keep its two fragments in VGPRs.
    const __bf16* gsrow = gsb + (size_t)bt * 64;
    FragA gsf[2];
    gsf[0] = load_row_frag(gsrow, 0, group);
    gsf[1] = load_row_frag(gsrow, 1, group);

    // B matrices for this gate (order pairs with plane order + gs).
    const size_t mats[7] = {
        (size_t)g,                 // U         <- p      (plane 0)
        9 + (size_t)g * 3 + 0,     // Wt[:64]   <- h_tb   (plane 1)
        9 + (size_t)g * 3 + 1,     // Wt[64:128]<- h      (plane 2)
        9 + (size_t)g * 3 + 2,     // Wt[128:]  <- h_ta   (plane 3)
        36 + (size_t)g,            // Ws        <- h_sb   (plane 4)
        45 + (size_t)g,            // Zt        <- g_t    (plane 5)
        54 + (size_t)g             // Zs_eff    <- g_s    (VGPR)
    };

    for (int nt = 0; nt < 4; ++nt) {
        v8f acc[4] = {v8f{}, v8f{}, v8f{}, v8f{}};
#pragma unroll
        for (int kstep = 0; kstep < 2; ++kstep) {
            v16bf bb[7];
#pragma unroll
            for (int i = 0; i < 7; ++i)
                bb[i] = load_frag_pk(spk, mats[i] * 8 + (size_t)(nt * 2 + kstep), lane);
#pragma unroll
            for (int mt = 0; mt < 4; ++mt) {
                int r = mt * 16 + m;
                FragA a0 = lds_row_frag(&apl[0][r][0], kstep, group);
                FragA a1 = lds_row_frag(&apl[1][r][0], kstep, group);
                FragA a2 = lds_row_frag(&apl[2][r][0], kstep, group);
                FragA a3 = lds_row_frag(&apl[3][r][0], kstep, group);
                FragA a4 = lds_row_frag(&apl[4][r][0], kstep, group);
                FragA a5 = lds_row_frag(&apl[5][r][0], kstep, group);
                acc[mt] = wmma_bf16(a0.v,        bb[0], acc[mt]);
                acc[mt] = wmma_bf16(a1.v,        bb[1], acc[mt]);
                acc[mt] = wmma_bf16(a2.v,        bb[2], acc[mt]);
                acc[mt] = wmma_bf16(a3.v,        bb[3], acc[mt]);
                acc[mt] = wmma_bf16(a4.v,        bb[4], acc[mt]);
                acc[mt] = wmma_bf16(a5.v,        bb[5], acc[mt]);
                acc[mt] = wmma_bf16(gsf[kstep].v, bb[6], acc[mt]);
            }
        }
        // C/D layout: VGPR r, lane L -> M = r + 8*(L>>4), N = L&15
#pragma unroll
        for (int mt = 0; mt < 4; ++mt)
#pragma unroll
            for (int r = 0; r < 8; ++r)
                pre_s[g][mt * 16 + r + 8 * group][lane & 15] = acc[mt][r];
        __syncthreads();

        // ---- Cell update for columns [nt*16, nt*16+16) ----
        for (int e = tid; e < 64 * 16; e += 288) {
            int rr = e >> 4;                 // bone 0..63
            int kk = nt * 16 + (e & 15);     // hidden col
            size_t row = (size_t)bt * 64 + rr;

            float pv[9];
#pragma unroll
            for (int gg = 0; gg < 9; ++gg)
                pv[gg] = pre_s[gg][rr][e & 15] + bmat[(size_t)gg * 4096 + rr * 64 + kk];

            float i_n  = sigm(pv[0]);
            float f_lt = sigm(pv[1]);
            float f_ft = sigm(pv[2]);
            float f_rt = sigm(pv[3]);
            float f_s  = sigm(pv[4]);
            float f_gt = sigm(pv[5]);
            float f_gs = sigm(pv[6]);
            float o_n  = sigm(pv[7]);
            float c_n  = tanhf(pv[8]);

            float c_old = c_in[row * 64 + kk];
            float ctb = (t > 0)      ? c_in[(row - 64) * 64 + kk] : 0.f;
            float cta = (t < T_ - 1) ? c_in[(row + 64) * 64 + kk] : 0.f;
            float csb = (rr > 0)     ? c_in[(row - 1) * 64 + kk]  : 0.f;
            float cgtv = cgt[((size_t)b * 64 + rr) * 64 + kk];
            float cgsv = cgs[(size_t)bt * 64 + kk];

            float cnew = f_lt * ctb + f_ft * c_old + f_rt * cta + f_s * csb +
                         f_gt * cgtv + f_gs * cgsv + c_n * i_n;
            float hnew = o_n * tanhf(cnew);

            if (c_out)    c_out[row * 64 + kk]   = cnew;
            if (hout_bf)  hout_bf[row * 64 + kk] = (__bf16)hnew;
            if (hout_f32) hout_f32[row * 64 + kk] = hnew;
        }
        __syncthreads();   // pre_s reused next nt
    }
}

// ---------------------------------------------------------------------------
// Host launcher
// ---------------------------------------------------------------------------
extern "C" void kernel_launch(void* const* d_in, const int* in_sizes, int n_in,
                              void* d_out, int out_size, void* d_ws, size_t ws_size,
                              hipStream_t stream) {
    (void)in_sizes; (void)n_in; (void)out_size; (void)ws_size;

    const float* enc  = (const float*)d_in[0];
    // d_in[1] = decoder_inputs (unused by the reference)
    const float* win  = (const float*)d_in[2];
    const float* bin  = (const float*)d_in[3];
    const float* U    = (const float*)d_in[4];
    const float* Wt   = (const float*)d_in[5];
    const float* Ws   = (const float*)d_in[6];
    const float* Zt   = (const float*)d_in[7];
    const float* Zs   = (const float*)d_in[8];
    const float* bmat = (const float*)d_in[9];
    float* out = (float*)d_out;

    char* ws = (char*)d_ws;
    size_t off = 0;
    auto take = [&](size_t bytes) -> void* {
        void* p = ws + off;
        off += (bytes + 255) & ~(size_t)255;
        return p;
    };

    float*          c0      = (float*)take(S_H * 4);
    float*          c1      = (float*)take(S_H * 4);
    __hip_bfloat16* pbf     = (__hip_bfloat16*)take(S_H * 2);
    __hip_bfloat16* hb1     = (__hip_bfloat16*)take(S_H * 2);
    __hip_bfloat16* hb2     = (__hip_bfloat16*)take(S_H * 2);
    __hip_bfloat16* encbf   = (__hip_bfloat16*)take(S_H * 2);
    __hip_bfloat16* winpk   = (__hip_bfloat16*)take((size_t)4096 * 4096 * 2);
    __hip_bfloat16* smallpk = (__hip_bfloat16*)take((size_t)63 * 4096 * 2);
    float*          cgt     = (float*)take((size_t)B_ * NB_ * H_ * 4);
    float*          cgs     = (float*)take((size_t)B_ * T_ * H_ * 4);
    __hip_bfloat16* gtb     = (__hip_bfloat16*)take((size_t)B_ * NB_ * H_ * 2);
    __hip_bfloat16* gsb     = (__hip_bfloat16*)take((size_t)B_ * T_ * H_ * 2);

    // 1) Pack weights into WMMA B-fragment layout (bf16)
    stln_pack_win  <<<4096, 256, 0, stream>>>(win, winpk);                 // 32768 waves
    stln_pack_small<<<  63, 256, 0, stream>>>(U, Wt, Ws, Zt, Zs, smallpk); // 504 waves

    // 2) Encoder -> bf16
    stln_cvt_bf16<<<(int)((S_H + 255) / 256), 256, 0, stream>>>(enc, encbf, (int)S_H);

    // 3) Input GEMM: h (f32 -> c0) and p (bf16 -> pbf).  16 Mblocks x 64 strips.
    stln_gemm_in<<<1024, 128, 0, stream>>>(encbf, winpk, bin, c0, pbf);

    // 4) Means (g_t, g_s) in f32 + bf16
    stln_mean_t<<<512, 256, 0, stream>>>(c0, cgt, gtb);
    stln_mean_s<<<504, 256, 0, stream>>>(c0, cgs, gsb);

    // 5) Three recurrent steps (ping-pong c and bf16 h buffers); 1 block/(b,t)
    stln_step<<<2016, 288, 0, stream>>>(pbf, pbf, c0, smallpk, bmat,
                                        gtb, gsb, cgt, cgs,
                                        c1, hb1, nullptr);
    stln_step<<<2016, 288, 0, stream>>>(pbf, hb1, c1, smallpk, bmat,
                                        gtb, gsb, cgt, cgs,
                                        c0, hb2, nullptr);
    stln_step<<<2016, 288, 0, stream>>>(pbf, hb2, c0, smallpk, bmat,
                                        gtb, gsb, cgt, cgs,
                                        nullptr, nullptr, out);
}
